// EdgeGNN_74826920231629
// MI455X (gfx1250) — matrix operand order repaired and checked by reading, not compiled, not run on hardware
//
#include <hip/hip_runtime.h>

// ---------------------------------------------------------------------------
// Types for CDNA5 WMMA (wave32, bf16 A/B, f32 accum)
// ---------------------------------------------------------------------------
typedef __attribute__((ext_vector_type(16))) __bf16        v16bf;
typedef __attribute__((ext_vector_type(8)))  float         v8f;
typedef __attribute__((ext_vector_type(8)))  unsigned int  v8u;

#define DMODEL 128
#define HID    256
#define MT     4          // 16-edge subtiles per block (64 edges/block)

struct GatherDesc {
    const float* base[6];
    int          sel[6];   // 0 = index by src[e], 1 = index by tgt[e], 2 = index by e
};

static __device__ __forceinline__ unsigned short f32_to_bf16(float f) {
    unsigned int u = __float_as_uint(f);
    u += 0x7FFFu + ((u >> 16) & 1u);          // round-to-nearest-even
    return (unsigned short)(u >> 16);
}

// A-matrix 16x32 bf16 layout (ISA 7.12.2): map k-in-chunk -> (lane half, element)
static __device__ __forceinline__ void a_pos(int kk, int& khalf, int& e) {
    khalf = (kk >> 3) & 1;
    e     = (kk & 7) + ((kk & 16) ? 8 : 0);
}

static __device__ __forceinline__ v8f wmma_bf16(v8u a_u, v8u b_u, v8f c) {
    v16bf a = __builtin_bit_cast(v16bf, a_u);
    v16bf b = __builtin_bit_cast(v16bf, b_u);
    return __builtin_amdgcn_wmma_f32_16x16x32_bf16(false, a, false, b, (short)0, c,
                                                   false, false);
}

// pack 4 consecutive-K f32 values (K multiple of 4) -> one 8-byte LDS store
static __device__ __forceinline__ void stage4(unsigned short* sA, int m, int k, float4 v) {
    int kc = k >> 5, kk = k & 31, khalf, el;
    a_pos(kk, khalf, el);                      // 4 consecutive k -> same lane, el..el+3
    unsigned int u0 = (unsigned int)f32_to_bf16(v.x) | ((unsigned int)f32_to_bf16(v.y) << 16);
    unsigned int u1 = (unsigned int)f32_to_bf16(v.z) | ((unsigned int)f32_to_bf16(v.w) << 16);
    *(uint2*)(sA + ((kc * 32 + khalf * 16 + m) << 4) + el) = make_uint2(u0, u1);
}

// ---------------------------------------------------------------------------
// Weight packing: W[K][Nout] f32 -> bf16 tiles in B-matrix VGPR layout.
// packed idx = ((ntile*KC + kc)*32 + lane)*16 + e ;  kk = (lane>>4)*8 + (e&7) + (e&8)*2
// ---------------------------------------------------------------------------
__global__ __launch_bounds__(256) void pack_b_kernel(const float* __restrict__ W,
                                                     unsigned short* __restrict__ dst,
                                                     int K, int Nout) {
    int idx = blockIdx.x * 256 + threadIdx.x;
    if (idx >= K * Nout) return;
    int e     = idx & 15;
    int lane  = (idx >> 4) & 31;
    int rest  = idx >> 9;
    int KC    = K >> 5;
    int kc    = rest % KC;
    int ntile = rest / KC;
    int kk    = ((lane >> 4) << 3) + (e & 7) + ((e & 8) << 1);
    int k     = (kc << 5) + kk;
    int n     = (ntile << 4) + (lane & 15);
    dst[idx]  = f32_to_bf16(W[(size_t)k * Nout + n]);
}

__global__ __launch_bounds__(256) void zero_kernel(float* __restrict__ p, int n) {
    int i = blockIdx.x * 256 + threadIdx.x;
    if (i < n) p[i] = 0.0f;
}

__global__ __launch_bounds__(256) void deg_kernel(const int* __restrict__ tgt,
                                                  float* __restrict__ deg, int E) {
    int i = blockIdx.x * 256 + threadIdx.x;
    if (i < E) unsafeAtomicAdd(&deg[tgt[i]], 1.0f);
}

__global__ __launch_bounds__(256) void invdeg_kernel(const float* __restrict__ deg,
                                                     float* __restrict__ inv, int N) {
    int i = blockIdx.x * 256 + threadIdx.x;
    if (i < N) inv[i] = 1.0f / fmaxf(deg[i], 1.0f);
}

// ---------------------------------------------------------------------------
// Fused 2-layer edge MLP on 64-edge blocks (4 x 16-edge WMMA subtiles):
//   K1 -> HID relu -> DMODEL
//   mode 0: out[e*128+n] = result      mode 1: atomic-add into agg[tgt[e]*128+n]
// ---------------------------------------------------------------------------
template <int K1>
__global__ __launch_bounds__(256) void edge_mlp_kernel(
    const int* __restrict__ src, const int* __restrict__ tgt, int E,
    GatherDesc g,
    const unsigned short* __restrict__ w1p, const float* __restrict__ b1,
    const unsigned short* __restrict__ w2p, const float* __restrict__ b2,
    float* __restrict__ out, float* __restrict__ agg, int mode) {

    constexpr int KC1 = K1 / 32;
    __shared__ union {
        unsigned short a[K1 * 16 * MT];        // A tiles (per subtile, A-layout)
        float          o[MT * 16 * DMODEL];    // f32 out tile (aliases dead A)
    } smA;
    __shared__ unsigned short sH[HID * 16 * MT];   // hidden tiles in A layout
    __shared__ int sSrc[MT * 16], sTgt[MT * 16];

    const int e0   = blockIdx.x * (MT * 16);
    const int tid  = threadIdx.x;
    const int wave = tid >> 5;
    const int lane = tid & 31;

    if (tid < MT * 16) {
        int e = e0 + tid; if (e >= E) e = E - 1;
        sSrc[tid] = src[e];
        sTgt[tid] = tgt[e];
    }
    __syncthreads();

    // ---- stage + convert A tiles (gathered concat features), float4 granular
    constexpr int KQ = K1 / 4;
    for (int idx = tid; idx < MT * 16 * KQ; idx += 256) {
        int m  = idx / KQ;                 // 0..63
        int k  = (idx - m * KQ) * 4;
        int seg = k >> 7, kk7 = k & 127;
        int sel = g.sel[seg];
        int row = (sel == 0) ? sSrc[m] : ((sel == 1) ? sTgt[m] : (e0 + m >= E ? E - 1 : e0 + m));
        float4 v = *(const float4*)(g.base[seg] + (size_t)row * DMODEL + kk7);
        stage4(smA.a + (m >> 4) * (K1 * 16), m & 15, k, v);
    }
    __syncthreads();

    // ---- layer 1: 16 column tiles of HID, 2 per wave, 4 subtiles -----------
    v8f accA[MT], accB[MT];
    #pragma unroll
    for (int st = 0; st < MT; ++st) { accA[st] = {}; accB[st] = {}; }
    for (int kc = 0; kc < KC1; ++kc) {
        v8u bu0 = *(const v8u*)(w1p + ((((size_t)wave       * KC1 + kc) * 32 + lane) << 4));
        v8u bu1 = *(const v8u*)(w1p + ((((size_t)(wave + 8) * KC1 + kc) * 32 + lane) << 4));
        #pragma unroll
        for (int st = 0; st < MT; ++st) {
            v8u au = *(const v8u*)(smA.a + st * (K1 * 16) + ((kc * 32 + lane) << 4));
            accA[st] = wmma_bf16(au, bu0, accA[st]);
            accB[st] = wmma_bf16(au, bu1, accB[st]);
        }
    }
    // epilogue: bias + relu, scatter hidden into LDS A-layout (K dim = column)
    {
        int nl = lane & 15, mh = (lane >> 4) << 3;
        #pragma unroll
        for (int t = 0; t < 2; ++t) {
            int n  = (wave + t * 8) * 16 + nl;
            float bv = b1[n];
            int kc = n >> 5, kk = n & 31, khalf, el;
            a_pos(kk, khalf, el);
            #pragma unroll
            for (int st = 0; st < MT; ++st) {
                v8f acc = t ? accB[st] : accA[st];
                #pragma unroll
                for (int r = 0; r < 8; ++r) {
                    int m = r + mh;
                    float v = fmaxf(acc[r] + bv, 0.0f);
                    sH[st * (HID * 16) + ((kc * 32 + khalf * 16 + m) << 4) + el] =
                        f32_to_bf16(v);
                }
            }
        }
    }
    __syncthreads();   // sA dead from here on; smA.o may be written after layer 2

    // ---- layer 2: K=256 (8 chunks), 8 column tiles of DMODEL, 1 per wave ---
    v8f acc2[MT];
    #pragma unroll
    for (int st = 0; st < MT; ++st) acc2[st] = {};
    for (int kc = 0; kc < 8; ++kc) {
        v8u bu = *(const v8u*)(w2p + ((((size_t)wave * 8 + kc) * 32 + lane) << 4));
        #pragma unroll
        for (int st = 0; st < MT; ++st) {
            v8u au = *(const v8u*)(sH + st * (HID * 16) + ((kc * 32 + lane) << 4));
            acc2[st] = wmma_bf16(au, bu, acc2[st]);
        }
    }
    // stage f32 out tile to LDS
    {
        int n = wave * 16 + (lane & 15);
        int mh = (lane >> 4) << 3;
        float bv = b2[n];
        #pragma unroll
        for (int st = 0; st < MT; ++st)
            #pragma unroll
            for (int r = 0; r < 8; ++r)
                smA.o[(st * 16 + r + mh) * DMODEL + n] = acc2[st][r] + bv;
    }
    __syncthreads();

    // ---- coalesced writeout / scatter --------------------------------------
    for (int idx = tid; idx < MT * 16 * (DMODEL / 4); idx += 256) {
        int m  = idx >> 5;                 // row within block (0..63)
        int c4 = (idx & 31) << 2;          // column (multiple of 4)
        int e  = e0 + m;
        if (e < E) {
            float4 v = *(const float4*)(smA.o + m * DMODEL + c4);
            if (mode == 0) {
                *(float4*)(out + (size_t)e * DMODEL + c4) = v;
            } else {
                float* dst = agg + (size_t)sTgt[m] * DMODEL + c4;
                unsafeAtomicAdd(dst + 0, v.x);
                unsafeAtomicAdd(dst + 1, v.y);
                unsafeAtomicAdd(dst + 2, v.z);
                unsafeAtomicAdd(dst + 3, v.w);
            }
        }
    }
}

// ---------------------------------------------------------------------------
// Node linear: out = (agg * inv_deg) @ W(128x128) + b      (16-node tiles)
// ---------------------------------------------------------------------------
__global__ __launch_bounds__(256) void node_linear_kernel(
    const float* __restrict__ agg, const float* __restrict__ inv, int N,
    const unsigned short* __restrict__ wp, const float* __restrict__ bias,
    float* __restrict__ out) {

    __shared__ unsigned short sA[DMODEL * 16];   // 4 chunks
    __shared__ float sO[16 * DMODEL];
    const int n0   = blockIdx.x * 16;
    const int tid  = threadIdx.x;
    const int wave = tid >> 5;
    const int lane = tid & 31;

    for (int idx = tid; idx < 16 * (DMODEL / 4); idx += 256) {
        int m = idx >> 5, k = (idx & 31) << 2;
        int row = n0 + m; if (row >= N) row = N - 1;
        float s = inv[row];
        float4 v = *(const float4*)(agg + (size_t)row * DMODEL + k);
        v.x *= s; v.y *= s; v.z *= s; v.w *= s;
        stage4(sA, m, k, v);
    }
    __syncthreads();

    v8f acc = {};
    #pragma unroll
    for (int kc = 0; kc < 4; ++kc) {
        v8u au = *(const v8u*)(sA + ((kc * 32 + lane) << 4));
        v8u bu = *(const v8u*)(wp + ((((size_t)wave * 4 + kc) * 32 + lane) << 4));
        acc = wmma_bf16(au, bu, acc);
    }
    {
        int n = wave * 16 + (lane & 15);
        int mh = (lane >> 4) << 3;
        float bv = bias[n];
        #pragma unroll
        for (int r = 0; r < 8; ++r)
            sO[(r + mh) * DMODEL + n] = acc[r] + bv;
    }
    __syncthreads();

    for (int idx = tid; idx < 16 * (DMODEL / 4); idx += 256) {
        int m = idx >> 5, c4 = (idx & 31) << 2;
        int row = n0 + m;
        if (row < N)
            *(float4*)(out + (size_t)row * DMODEL + c4) = *(const float4*)(sO + m * DMODEL + c4);
    }
}

// ---------------------------------------------------------------------------
// out = LayerNorm(A + B) * gamma + beta   (wave per row, D=128)
// ---------------------------------------------------------------------------
__global__ __launch_bounds__(256) void ln_residual_kernel(
    const float* __restrict__ A, const float* __restrict__ B,
    const float* __restrict__ gamma, const float* __restrict__ beta,
    float* __restrict__ out, int rows) {

    int lane = threadIdx.x & 31;
    int row  = blockIdx.x * 8 + (threadIdx.x >> 5);
    if (row >= rows) return;

    float4 va = ((const float4*)(A + (size_t)row * DMODEL))[lane];
    float4 vb = ((const float4*)(B + (size_t)row * DMODEL))[lane];
    float vx = va.x + vb.x, vy = va.y + vb.y, vz = va.z + vb.z, vw = va.w + vb.w;

    float s = vx + vy + vz + vw;
    for (int o = 16; o; o >>= 1) s += __shfl_xor(s, o, 32);
    float mu = s * (1.0f / 128.0f);
    float dx = vx - mu, dy = vy - mu, dz = vz - mu, dw = vw - mu;
    float q = dx * dx + dy * dy + dz * dz + dw * dw;
    for (int o = 16; o; o >>= 1) q += __shfl_xor(q, o, 32);
    float rstd = rsqrtf(q * (1.0f / 128.0f) + 1e-5f);

    float4 g4 = ((const float4*)gamma)[lane];
    float4 b4 = ((const float4*)beta)[lane];
    float4 o4 = { dx * rstd * g4.x + b4.x, dy * rstd * g4.y + b4.y,
                  dz * rstd * g4.z + b4.z, dw * rstd * g4.w + b4.w };
    ((float4*)(out + (size_t)row * DMODEL))[lane] = o4;
}

// ---------------------------------------------------------------------------
// Host orchestration
// ---------------------------------------------------------------------------
extern "C" void kernel_launch(void* const* d_in, const int* in_sizes, int n_in,
                              void* d_out, int out_size, void* d_ws, size_t ws_size,
                              hipStream_t stream) {
    const float* x         = (const float*)d_in[0];
    const float* edge_attr = (const float*)d_in[1];
    const int*   ei        = (const int*)d_in[2];
    const float* enc_w1 = (const float*)d_in[3];  const float* enc_b1 = (const float*)d_in[4];
    const float* enc_w2 = (const float*)d_in[5];  const float* enc_b2 = (const float*)d_in[6];
    const float* em_w1  = (const float*)d_in[7];  const float* em_b1  = (const float*)d_in[8];
    const float* em_w2  = (const float*)d_in[9];  const float* em_b2  = (const float*)d_in[10];
    const float* nm_w1  = (const float*)d_in[11]; const float* nm_b1  = (const float*)d_in[12];
    const float* nm_w2  = (const float*)d_in[13]; const float* nm_b2  = (const float*)d_in[14];
    const float* no_w   = (const float*)d_in[15]; const float* no_b   = (const float*)d_in[16];
    const float* node_ln_g = (const float*)d_in[17]; const float* node_ln_b = (const float*)d_in[18];
    const float* edge_ln_g = (const float*)d_in[19]; const float* edge_ln_b = (const float*)d_in[20];

    const int N = in_sizes[0] / DMODEL;
    const int E = in_sizes[1] / DMODEL;
    const int* srcI = ei;
    const int* tgtI = ei + E;

    // workspace carve-out
    char* base = (char*)d_ws;
    size_t off = 0;
    auto carve = [&](size_t bytes) { char* p = base + off; off += (bytes + 255) & ~(size_t)255; return p; };
    float* init_edge = (float*)carve((size_t)E * DMODEL * 4);
    float* edge_raw  = (float*)carve((size_t)E * DMODEL * 4);
    float* agg       = (float*)carve((size_t)N * DMODEL * 4);
    float* node_tmp  = (float*)carve((size_t)N * DMODEL * 4);
    float* deg       = (float*)carve((size_t)N * 4);
    float* invd      = (float*)carve((size_t)N * 4);
    unsigned short* enc1p = (unsigned short*)carve((size_t)384 * 256 * 2);
    unsigned short* enc2p = (unsigned short*)carve((size_t)256 * 128 * 2);
    unsigned short* em1p  = (unsigned short*)carve((size_t)768 * 256 * 2);
    unsigned short* em2p  = (unsigned short*)carve((size_t)256 * 128 * 2);
    unsigned short* nm1p  = (unsigned short*)carve((size_t)384 * 256 * 2);
    unsigned short* nm2p  = (unsigned short*)carve((size_t)256 * 128 * 2);
    unsigned short* nowp  = (unsigned short*)carve((size_t)128 * 128 * 2);

    float* node_f_out = (float*)d_out;
    float* edge_f_out = node_f_out + (size_t)N * DMODEL;

    auto pack = [&](const float* W, unsigned short* dst, int K, int Nout) {
        pack_b_kernel<<<(K * Nout + 255) / 256, 256, 0, stream>>>(W, dst, K, Nout);
    };
    pack(enc_w1, enc1p, 384, 256); pack(enc_w2, enc2p, 256, 128);
    pack(em_w1,  em1p,  768, 256); pack(em_w2,  em2p,  256, 128);
    pack(nm_w1,  nm1p,  384, 256); pack(nm_w2,  nm2p,  256, 128);
    pack(no_w,   nowp,  128, 128);

    const int edgeBlocks = (E + MT * 16 - 1) / (MT * 16);
    const int nodeTiles  = (N + 15) / 16;

    // ---- edge encoder: cat[x[src], x[tgt], edge_attr] -> init_edge ---------
    GatherDesc ge = {};
    ge.base[0] = x; ge.sel[0] = 0;
    ge.base[1] = x; ge.sel[1] = 1;
    ge.base[2] = edge_attr; ge.sel[2] = 2;
    edge_mlp_kernel<384><<<edgeBlocks, 256, 0, stream>>>(srcI, tgtI, E, ge,
        enc1p, enc_b1, enc2p, enc_b2, init_edge, agg, 0);

    // ---- degrees -----------------------------------------------------------
    zero_kernel<<<(N + 255) / 256, 256, 0, stream>>>(deg, N);
    deg_kernel<<<(E + 255) / 256, 256, 0, stream>>>(tgtI, deg, E);
    invdeg_kernel<<<(N + 255) / 256, 256, 0, stream>>>(deg, invd, N);

    const float* node_cur = x;
    const float* edge_cur = init_edge;

    for (int it = 0; it < 3; ++it) {
        // edge update: cat[x[src], nf[src], x[tgt], nf[tgt], init_edge, ef]
        GatherDesc gu = {};
        gu.base[0] = x;         gu.sel[0] = 0;
        gu.base[1] = node_cur;  gu.sel[1] = 0;
        gu.base[2] = x;         gu.sel[2] = 1;
        gu.base[3] = node_cur;  gu.sel[3] = 1;
        gu.base[4] = init_edge; gu.sel[4] = 2;
        gu.base[5] = edge_cur;  gu.sel[5] = 2;
        edge_mlp_kernel<768><<<edgeBlocks, 256, 0, stream>>>(srcI, tgtI, E, gu,
            em1p, em_b1, em2p, em_b2, edge_raw, agg, 0);

        // message MLP: cat[x[src], nf[src], edge_raw] -> scatter-add agg[tgt]
        zero_kernel<<<(N * DMODEL + 255) / 256, 256, 0, stream>>>(agg, N * DMODEL);
        GatherDesc gm = {};
        gm.base[0] = x;        gm.sel[0] = 0;
        gm.base[1] = node_cur; gm.sel[1] = 0;
        gm.base[2] = edge_raw; gm.sel[2] = 2;
        edge_mlp_kernel<384><<<edgeBlocks, 256, 0, stream>>>(srcI, tgtI, E, gm,
            nm1p, nm_b1, nm2p, nm_b2, nullptr, agg, 1);

        // node linear on scaled aggregate, then residual + LN
        node_linear_kernel<<<nodeTiles, 256, 0, stream>>>(agg, invd, N, nowp, no_b, node_tmp);
        ln_residual_kernel<<<(N + 7) / 8, 256, 0, stream>>>(x, node_tmp,
            node_ln_g, node_ln_b, node_f_out, N);
        ln_residual_kernel<<<(E + 7) / 8, 256, 0, stream>>>(init_edge, edge_raw,
            edge_ln_g, edge_ln_b, edge_f_out, E);

        node_cur = node_f_out;
        edge_cur = edge_f_out;
    }
}
